// QSCrossAttention_45612552684055
// MI455X (gfx1250) — matrix operand order repaired and checked by reading, not compiled, not run on hardware
//
#include <hip/hip_runtime.h>
#include <hip/hip_bf16.h>
#include <stdint.h>

#define DIM 512
#define BATCH 8
#define NQ 1024
#define NKV 4096

#define BM 128
#define BN 64
#define BK 32
#define LDSP 40   // LDS row pitch in bf16 elements (80B) via TDM padding

typedef __bf16 bf16_t;
typedef __attribute__((ext_vector_type(16))) __bf16 v16bf;
typedef __attribute__((ext_vector_type(8)))  float  v8f;
typedef __attribute__((ext_vector_type(4)))  unsigned int v4u;
typedef __attribute__((ext_vector_type(8)))  int v8i;
typedef __attribute__((ext_vector_type(4)))  int v4i;

union FragAB { v16bf v; uint4 q[2]; };

__device__ __forceinline__ bf16_t f2bf(float f) {
  uint32_t u = __builtin_bit_cast(uint32_t, f);
  uint32_t r = u + 0x7FFFu + ((u >> 16) & 1u);   // round-to-nearest-even
  uint16_t h = (uint16_t)(r >> 16);
  return __builtin_bit_cast(bf16_t, h);
}
__device__ __forceinline__ float bf2f(bf16_t b) {
  uint16_t h = __builtin_bit_cast(uint16_t, b);
  uint32_t u = ((uint32_t)h) << 16;
  return __builtin_bit_cast(float, u);
}

__device__ __forceinline__ uint32_t lds_off_u32(const void* p) {
  // addrspacecast(LDS->flat) keeps the workgroup-relative LDS byte offset in
  // the low 32 bits (flat aperture encodes LDS offset in addr[31:0]).
  return (uint32_t)(uintptr_t)p;
}

// TDM: DMA a 2D tile (32 bf16 per row x `rows` rows) from global to LDS.
// LDS padding: +4 DWORDs after every 16 DWORDs -> effective 40-elem row pitch.
// Descriptor bitfields per CDNA5 ISA 8.3/8.4 (D# group0 + group1; 2D so
// groups 2/3 are zero and tile_dim2=0 marks higher dims unused).
__device__ __forceinline__ void tdm_load_tile_2d(uint32_t lds_byte_addr,
                                                 const bf16_t* gsrc,
                                                 uint32_t rows,
                                                 uint32_t stride_elems) {
  uint64_t ga = (uint64_t)(uintptr_t)gsrc;
  v4u g0;
  g0[0] = 1u;                                    // count=1 (valid user D#)
  g0[1] = lds_byte_addr;                         // bits 63:32  lds_addr
  g0[2] = (uint32_t)ga;                          // bits 95:64  global_addr lo
  g0[3] = (uint32_t)((ga >> 32) & 0x01FFFFFFu)   // bits 120:96 global_addr hi
        | (2u << 30);                            // bits 127:126 type=2 (image)

  const uint32_t tdim0 = 32u, til0 = 32u;        // row length (elements)
  const uint32_t tdim1 = rows, til1 = rows;
  v8i g1;
  g1[0] = (int)((1u << 16)      // data_size = 2 bytes
              | (1u << 20)      // pad_enable
              | (3u << 22)      // pad_interval: every 16 DWORDs (64B)
              | (3u << 25));    // pad_amount: 4 DWORDs (16B)
  g1[1] = (int)((tdim0 & 0xFFFFu) << 16);                               // tensor_dim0 lo
  g1[2] = (int)(((tdim0 >> 16) & 0xFFFFu) | ((tdim1 & 0xFFFFu) << 16)); // dim0 hi | dim1 lo
  g1[3] = (int)(((tdim1 >> 16) & 0xFFFFu) | ((til0 & 0xFFFFu) << 16));  // dim1 hi | tile_dim0
  g1[4] = (int)(til1 & 0xFFFFu);                 // tile_dim1 (tile_dim2 = 0)
  g1[5] = (int)stride_elems;                     // tensor_dim0_stride lo32
  g1[6] = 0;                                     // stride0 hi16 | stride1 lo16
  g1[7] = 0;                                     // stride1 hi32
  v4i z4 = {0, 0, 0, 0};
  v8i z8 = {0, 0, 0, 0, 0, 0, 0, 0};
  // clang-23 / therock lane: 6-arg form (g0, g1, g2, g3, extra, cpol)
  __builtin_amdgcn_tensor_load_to_lds(g0, g1, z4, z4, z8, 0);
}

// ---------------------------------------------------------------- convert ---
__global__ void cvt_f32_bf16(const float* __restrict__ in,
                             bf16_t* __restrict__ out, int n) {
  int i = blockIdx.x * blockDim.x + threadIdx.x;
  if (i < n) out[i] = f2bf(in[i]);
}

// ------------------------------------------------------------------- GEMM ---
// C[M,N] = A[M,K] * Bt[N,K]^T.  Block: 128 threads (4 waves), 128x64 C tile,
// BK=32, double-buffered LDS staged by the Tensor Data Mover (wave 0 issues,
// TENSORcnt throttles a 2-deep pipeline). Wave w owns rows [w*32, w*32+32)
// x all 64 columns -> 8 WMMAs per K-step per wave.
__global__ __launch_bounds__(128) void gemm_bf16_wmma(
    const bf16_t* __restrict__ A,  int lda, long long bsA,
    const bf16_t* __restrict__ Bt, int ldb, long long bsB,
    void* __restrict__ Cv,         int ldc, long long bsC,
    int K, int out_f32)
{
  __shared__ __align__(16) bf16_t sA[2][BM * LDSP];
  __shared__ __align__(16) bf16_t sB[2][BN * LDSP];

  const int b = blockIdx.z;
  A  += (size_t)b * (size_t)bsA;
  Bt += (size_t)b * (size_t)bsB;

  const int n0   = blockIdx.x * BN;
  const int m0   = blockIdx.y * BM;
  const int t    = threadIdx.x;
  const int lane = t & 31;
  const int w    = t >> 5;
  const int r    = lane & 15;
  const int hi   = lane >> 4;

  v8f acc[2][4] = {};

  const int ksteps = K >> 5;

  if (w == 0) {  // prologue: fill buffer 0
    tdm_load_tile_2d(lds_off_u32(&sA[0][0]), A + (size_t)m0 * lda, BM, (uint32_t)lda);
    tdm_load_tile_2d(lds_off_u32(&sB[0][0]), Bt + (size_t)n0 * ldb, BN, (uint32_t)ldb);
  }

  for (int kt = 0; kt < ksteps; ++kt) {
    const int cur = kt & 1;
    if (w == 0) {
      if (kt + 1 < ksteps) {   // stage next tile into the other buffer
        const int k1 = (kt + 1) << 5;
        tdm_load_tile_2d(lds_off_u32(&sA[cur ^ 1][0]),
                         A + (size_t)m0 * lda + k1, BM, (uint32_t)lda);
        tdm_load_tile_2d(lds_off_u32(&sB[cur ^ 1][0]),
                         Bt + (size_t)n0 * ldb + k1, BN, (uint32_t)ldb);
        __builtin_amdgcn_s_wait_tensorcnt(2);  // oldest 2 (current buf) done
      } else {
        __builtin_amdgcn_s_wait_tensorcnt(0);
      }
    }
    __syncthreads();

    // A fragments: lane holds row; elements 0-7 = K kbase..kbase+7,
    // elements 8-15 = K 16+kbase.. (kbase = hi*8) -> two 16B LDS loads
    FragAB a[2];
    #pragma unroll
    for (int st = 0; st < 2; ++st) {
      const bf16_t* arow = &sA[cur][(w * 32 + st * 16 + r) * LDSP];
      a[st].q[0] = *(const uint4*)(arow + hi * 8);
      a[st].q[1] = *(const uint4*)(arow + 16 + hi * 8);
    }
    #pragma unroll
    for (int nt = 0; nt < 4; ++nt) {
      // B fragment: lane holds column; elements 0-7 = K hi*8.., 8-15 = K 16+hi*8..
      FragAB bb;
      const bf16_t* brow = &sB[cur][(nt * 16 + r) * LDSP];
      bb.q[0] = *(const uint4*)(brow + hi * 8);
      bb.q[1] = *(const uint4*)(brow + 16 + hi * 8);
      #pragma unroll
      for (int st = 0; st < 2; ++st) {
        acc[st][nt] = __builtin_amdgcn_wmma_f32_16x16x32_bf16(
            false, a[st].v, false, bb.v, (short)0, acc[st][nt], false, false);
      }
    }
    __syncthreads();
  }

  // Epilogue: C element e -> (M = e + hi*8, N = lane&15) within a 16x16 tile
  #pragma unroll
  for (int st = 0; st < 2; ++st) {
    const int mrow = m0 + w * 32 + st * 16 + hi * 8;
    #pragma unroll
    for (int nt = 0; nt < 4; ++nt) {
      const int n = n0 + nt * 16 + r;
      #pragma unroll
      for (int e = 0; e < 8; ++e) {
        const size_t ci = (size_t)b * (size_t)bsC + (size_t)(mrow + e) * ldc + n;
        if (out_f32) ((float*)Cv)[ci] = acc[st][nt][e];
        else         ((bf16_t*)Cv)[ci] = f2bf(acc[st][nt][e]);
      }
    }
  }
}

// ---------------------------------------------------------------- softmax ---
__global__ __launch_bounds__(256) void softmax_mask_kernel(
    const bf16_t* __restrict__ S, const uint8_t* __restrict__ mask,
    bf16_t* __restrict__ P)
{
  const int n = blockIdx.x;
  const int b = blockIdx.y;
  const bf16_t* srow = S + ((size_t)b * NQ + n) * NKV;
  bf16_t*       prow = P + ((size_t)b * NQ + n) * NKV;
  const uint8_t* mrow = mask + (size_t)b * NKV;
  const float scale = 0.04419417382415922f;   // 512^-0.5

  float x[16];
  float mx = -1e30f;
  #pragma unroll
  for (int i = 0; i < 16; ++i) {
    const int idx = threadIdx.x + (i << 8);
    float s = bf2f(srow[idx]) * scale + (mrow[idx] ? 0.0f : -10000.0f);
    x[i] = s;
    mx = fmaxf(mx, s);
  }

  __shared__ float rmax[8];
  __shared__ float rsum[8];
  #pragma unroll
  for (int o = 16; o > 0; o >>= 1) mx = fmaxf(mx, __shfl_xor(mx, o));
  if ((threadIdx.x & 31) == 0) rmax[threadIdx.x >> 5] = mx;
  __syncthreads();
  if (threadIdx.x == 0) {
    float m = rmax[0];
    for (int i = 1; i < 8; ++i) m = fmaxf(m, rmax[i]);
    rmax[0] = m;
  }
  __syncthreads();
  mx = rmax[0];

  float sum = 0.0f;
  #pragma unroll
  for (int i = 0; i < 16; ++i) { x[i] = __expf(x[i] - mx); sum += x[i]; }
  #pragma unroll
  for (int o = 16; o > 0; o >>= 1) sum += __shfl_xor(sum, o);
  if ((threadIdx.x & 31) == 0) rsum[threadIdx.x >> 5] = sum;
  __syncthreads();
  if (threadIdx.x == 0) {
    float s = 0.0f;
    for (int i = 0; i < 8; ++i) s += rsum[i];
    rsum[0] = s;
  }
  __syncthreads();
  const float inv = 1.0f / rsum[0];

  #pragma unroll
  for (int i = 0; i < 16; ++i) {
    const int idx = threadIdx.x + (i << 8);
    prow[idx] = f2bf(x[i] * inv);
  }
}

// ----------------------------------------------------------------- launch ---
extern "C" void kernel_launch(void* const* d_in, const int* in_sizes, int n_in,
                              void* d_out, int out_size, void* d_ws, size_t ws_size,
                              hipStream_t stream) {
  (void)in_sizes; (void)n_in; (void)out_size; (void)ws_size;

  const float*   prototype = (const float*)d_in[0];
  const float*   q_x       = (const float*)d_in[1];
  const uint8_t* mask      = (const uint8_t*)d_in[2];
  const float*   Wq        = (const float*)d_in[3];
  const float*   Wk        = (const float*)d_in[4];
  const float*   Wv        = (const float*)d_in[5];
  const float*   Wp        = (const float*)d_in[6];

  char* ws = (char*)d_ws;
  size_t off = 0;
  auto alloc = [&](size_t elems) -> bf16_t* {
    bf16_t* p = (bf16_t*)(ws + off);
    off += (elems * 2 + 255) & ~(size_t)255;
    return p;
  };
  bf16_t* protoB = alloc((size_t)BATCH * NQ  * DIM);  // bf16 prototype
  bf16_t* qxB    = alloc((size_t)BATCH * NKV * DIM);  // bf16 q_x
  bf16_t* WqB    = alloc((size_t)DIM * DIM);
  bf16_t* WkB    = alloc((size_t)DIM * DIM);
  bf16_t* WvB    = alloc((size_t)DIM * DIM);
  bf16_t* WpB    = alloc((size_t)DIM * DIM);
  bf16_t* qB     = alloc((size_t)BATCH * NQ  * DIM);  // q  = proto @ Wq^T
  bf16_t* kB     = alloc((size_t)BATCH * NKV * DIM);  // k  = q_x @ Wk^T
  bf16_t* vTB    = alloc((size_t)BATCH * DIM * NKV);  // vT = Wv @ q_x^T
  bf16_t* Sb     = alloc((size_t)BATCH * NQ  * NKV);  // logits (unscaled)
  bf16_t* Pb     = alloc((size_t)BATCH * NQ  * NKV);  // softmax probs
  bf16_t* xB     = alloc((size_t)BATCH * NQ  * DIM);  // attn output

  auto cvt = [&](const float* in, bf16_t* out, size_t n) {
    cvt_f32_bf16<<<dim3((unsigned)((n + 255) / 256)), 256, 0, stream>>>(in, out, (int)n);
  };
  cvt(prototype, protoB, (size_t)BATCH * NQ  * DIM);
  cvt(q_x,       qxB,    (size_t)BATCH * NKV * DIM);
  cvt(Wq, WqB, (size_t)DIM * DIM);
  cvt(Wk, WkB, (size_t)DIM * DIM);
  cvt(Wv, WvB, (size_t)DIM * DIM);
  cvt(Wp, WpB, (size_t)DIM * DIM);

  const long long sPQ = (long long)NQ  * DIM;
  const long long sKV = (long long)NKV * DIM;
  const long long sS  = (long long)NQ  * NKV;

  // q[b] = protoB[b] @ WqB^T     (M=NQ, N=DIM, K=DIM)
  gemm_bf16_wmma<<<dim3(DIM / BN, NQ / BM, BATCH), 128, 0, stream>>>(
      protoB, DIM, sPQ, WqB, DIM, 0, qB, DIM, sPQ, DIM, 0);
  // k[b] = qxB[b] @ WkB^T        (M=NKV, N=DIM, K=DIM)
  gemm_bf16_wmma<<<dim3(DIM / BN, NKV / BM, BATCH), 128, 0, stream>>>(
      qxB, DIM, sKV, WkB, DIM, 0, kB, DIM, sKV, DIM, 0);
  // vT[b] = WvB @ qxB[b]^T       (M=DIM, N=NKV, K=DIM)
  gemm_bf16_wmma<<<dim3(NKV / BN, DIM / BM, BATCH), 128, 0, stream>>>(
      WvB, DIM, 0, qxB, DIM, sKV, vTB, NKV, sKV, DIM, 0);
  // S[b] = qB[b] @ kB[b]^T       (M=NQ, N=NKV, K=DIM)
  gemm_bf16_wmma<<<dim3(NKV / BN, NQ / BM, BATCH), 128, 0, stream>>>(
      qB, DIM, sPQ, kB, DIM, sKV, Sb, NKV, sS, DIM, 0);
  // P = softmax(S * scale + mask_bias)
  softmax_mask_kernel<<<dim3(NQ, BATCH), 256, 0, stream>>>(Sb, mask, Pb);
  // x[b] = P[b] @ v[b]  (= P @ (vT)^T)   (M=NQ, N=DIM, K=NKV)
  gemm_bf16_wmma<<<dim3(DIM / BN, NQ / BM, BATCH), 128, 0, stream>>>(
      Pb, NKV, sS, vTB, NKV, sKV, xB, DIM, sPQ, NKV, 0);
  // out[b] = x[b] @ Wp^T  -> f32  (M=NQ, N=DIM, K=DIM)
  gemm_bf16_wmma<<<dim3(DIM / BN, NQ / BM, BATCH), 128, 0, stream>>>(
      xB, DIM, sPQ, WpB, DIM, 0, d_out, DIM, sPQ, DIM, 1);
}
